// DenseDeepGCN_27616639713516
// MI455X (gfx1250) — matrix operand order repaired and checked by reading, not compiled, not run on hardware
//
#include <hip/hip_runtime.h>

// ============================================================================
// DenseDeepGCN forward for MI455X (gfx1250, wave32, WMMA).
// All 1x1 convs run as f16 WMMA GEMMs (v_wmma_f32_16x16x32_f16, f32 accum).
// Weight tiles are staged in LDS via GLOBAL_LOAD_ASYNC_TO_LDS (ASYNCcnt) when
// the toolchain exposes the builtin. KNN / BN stats / reductions stay fp32.
// BN uses batch statistics, so conv biases cancel; final conv keeps bias.
// ============================================================================

typedef __attribute__((ext_vector_type(16))) _Float16 v16h;
typedef __attribute__((ext_vector_type(8)))  float    v8f;
typedef __attribute__((ext_vector_type(4)))  int      v4i;

#define BN_EPS 1e-5f

static constexpr int BB   = 2;      // batch
static constexpr int NP   = 8192;   // points
static constexpr int MT   = BB*NP;  // 16384
static constexpr int KNN  = 5;      // neighbors kept

#if __has_builtin(__builtin_amdgcn_global_load_async_to_lds_b128)
#define HAVE_ASYNC_LDS 1
#endif
#if __has_builtin(__builtin_amdgcn_s_wait_asynccnt)
#define HAVE_WAIT_ASYNC 1
#endif

// ---------------------------------------------------------------------------
// WMMA GEMM:  Y[O, Mc] = W[O, C] * X[Mc, C]^T    (f16 in, f32 out)
// C multiple of 32, O multiple of 16, Mc multiple of 512.
// Block = 256 threads = 8 waves. The block's 16 weight rows are staged in LDS
// (async-to-LDS path when available); each wave owns a 16x64 output tile
// (4 independent accumulators -> 4 WMMAs per K-step, hazard-free pipelining).
//
// A (16x32, f16): lane l holds row m=l&15; K = k0..k0+15,
//   k0 = (l>=16)?16:0  -> two contiguous uint4 reads (from LDS tile).
// B (32x16, f16): lane l holds col n=l&15; K = kb..kb+15, kb=(l>=16)?16:0
//   -> two contiguous uint4 loads from row-major X (point-major features).
// D (16x16, f32): VGPR v: lanes0-15 -> M=v, lanes16-31 -> M=v+8; N = l&15.
// ---------------------------------------------------------------------------
__global__ void wmma_gemm_f16(const _Float16* __restrict__ W,
                              const _Float16* __restrict__ X,
                              float* __restrict__ Y,
                              int O, int C, int Mc) {
  extern __shared__ _Float16 sW[];            // 16 * C halves
  union AB { uint4 q[2]; v16h v; };
  union CF { v8f v; float f[8]; };

  const int rowBase = blockIdx.y * 16;

  // ---- stage 16 x C weight tile into LDS (b128 chunks of 8 halves) ----
  const int cch = C >> 3;                     // chunks per row
  for (int i = threadIdx.x; i < 16 * cch; i += 256) {
    const int r  = i / cch;
    const int cc = (i % cch) << 3;
    const _Float16* gp = W + (size_t)(rowBase + r) * C + cc;
    _Float16*       lp = sW + r * C + cc;
#if HAVE_ASYNC_LDS
    // Builtin signature (per hipcc diagnostic): AS1 v4i* src, AS3 v4i* dst.
    __builtin_amdgcn_global_load_async_to_lds_b128(
        (__attribute__((address_space(1))) v4i*)(void*)gp,
        (__attribute__((address_space(3))) v4i*)(void*)lp, 0, 0);
#else
    *reinterpret_cast<uint4*>(lp) = *reinterpret_cast<const uint4*>(gp);
#endif
  }
#if HAVE_ASYNC_LDS
#if HAVE_WAIT_ASYNC
  __builtin_amdgcn_s_wait_asynccnt(0);
#else
  asm volatile("s_wait_asynccnt 0" ::: "memory");
#endif
#endif
  __syncthreads();

  const int lane    = threadIdx.x & 31;
  const int wave    = threadIdx.x >> 5;
  const int half    = lane >> 4;
  const int kA0     = half << 4;              // 0 or 16
  const int kB0     = half << 4;              // 0 or 16
  const int colBase = (blockIdx.x * 8 + wave) * 64;

  const _Float16* aRow = sW + (lane & 15) * C;
  const _Float16* xr0  = X + (size_t)(colBase +  0 + (lane & 15)) * C;
  const _Float16* xr1  = X + (size_t)(colBase + 16 + (lane & 15)) * C;
  const _Float16* xr2  = X + (size_t)(colBase + 32 + (lane & 15)) * C;
  const _Float16* xr3  = X + (size_t)(colBase + 48 + (lane & 15)) * C;

  CF acc[4];
  #pragma unroll
  for (int t = 0; t < 4; ++t)
    #pragma unroll
    for (int i = 0; i < 8; ++i) acc[t].f[i] = 0.0f;

  for (int kk = 0; kk < C; kk += 32) {
    if (kk + 32 < C) {                        // hint next K tile into cache
      __builtin_prefetch(xr0 + kk + 32, 0, 3);
      __builtin_prefetch(xr1 + kk + 32, 0, 3);
      __builtin_prefetch(xr2 + kk + 32, 0, 3);
      __builtin_prefetch(xr3 + kk + 32, 0, 3);
    }
    AB a;
    a.q[0] = *reinterpret_cast<const uint4*>(aRow + kk + kA0);
    a.q[1] = *reinterpret_cast<const uint4*>(aRow + kk + kA0 + 8);

    AB b0, b1, b2, b3;
    b0.q[0] = *reinterpret_cast<const uint4*>(xr0 + kk + kB0);
    b0.q[1] = *reinterpret_cast<const uint4*>(xr0 + kk + kB0 + 8);
    b1.q[0] = *reinterpret_cast<const uint4*>(xr1 + kk + kB0);
    b1.q[1] = *reinterpret_cast<const uint4*>(xr1 + kk + kB0 + 8);
    b2.q[0] = *reinterpret_cast<const uint4*>(xr2 + kk + kB0);
    b2.q[1] = *reinterpret_cast<const uint4*>(xr2 + kk + kB0 + 8);
    b3.q[0] = *reinterpret_cast<const uint4*>(xr3 + kk + kB0);
    b3.q[1] = *reinterpret_cast<const uint4*>(xr3 + kk + kB0 + 8);

    acc[0].v = __builtin_amdgcn_wmma_f32_16x16x32_f16(false, a.v, false, b0.v,
                                                      (short)0, acc[0].v, false, false);
    acc[1].v = __builtin_amdgcn_wmma_f32_16x16x32_f16(false, a.v, false, b1.v,
                                                      (short)0, acc[1].v, false, false);
    acc[2].v = __builtin_amdgcn_wmma_f32_16x16x32_f16(false, a.v, false, b2.v,
                                                      (short)0, acc[2].v, false, false);
    acc[3].v = __builtin_amdgcn_wmma_f32_16x16x32_f16(false, a.v, false, b3.v,
                                                      (short)0, acc[3].v, false, false);
  }

  #pragma unroll
  for (int t = 0; t < 4; ++t)
    #pragma unroll
    for (int v = 0; v < 8; ++v)
      Y[(size_t)(rowBase + v + half * 8) * Mc + colBase + t * 16 + (lane & 15)] =
          acc[t].f[v];
}

// ---------------------------------------------------------------------------
// Brute-force KNN with LDS tiling. x: [B*N, C] point-major fp32.
// Each thread owns one query, keeps a sorted top-KD (dist,idx) list in
// registers (fully unrolled insertion -> no scratch), applies dilation at
// writeout: idxOut[m*k + i] = list[i*dil].
// ---------------------------------------------------------------------------
template<int KD, int CMAX>
__global__ void knn_kernel(const float* __restrict__ x,
                           int* __restrict__ idxOut,
                           int C, int Nn, int kOut, int dil) {
  __shared__ float sQ[64 * CMAX];
  __shared__ float sP[32 * CMAX];
  const int t = threadIdx.x;                 // 0..63
  const int b = blockIdx.y;
  const int q = blockIdx.x * 64 + t;
  const int m = b * Nn + q;

  for (int c = 0; c < C; ++c) sQ[t * C + c] = x[(size_t)m * C + c];

  float bd[KD]; int bi[KD];
  #pragma unroll
  for (int i = 0; i < KD; ++i) { bd[i] = 3.4e38f; bi[i] = 0; }

  for (int j0 = 0; j0 < Nn; j0 += 32) {
    __syncthreads();
    for (int i = t; i < 32 * C; i += 64)
      sP[i] = x[(size_t)(b * Nn + j0 + i / C) * C + (i % C)];
    __syncthreads();
    for (int jj = 0; jj < 32; ++jj) {
      float d = 0.f;
      for (int c = 0; c < C; ++c) {
        float df = sQ[t * C + c] - sP[jj * C + c];
        d += df * df;
      }
      const int j = j0 + jj;
      if (d < bd[KD - 1]) {
        #pragma unroll
        for (int p = KD - 1; p >= 0; --p) {
          const float prev  = (p > 0) ? bd[p - 1] : -3.4e38f;
          const int   prevI = (p > 0) ? bi[p - 1] : 0;
          if (d < bd[p]) {
            if (d < prev) { bd[p] = prev; bi[p] = prevI; }
            else          { bd[p] = d;    bi[p] = j;     }
          }
        }
      }
    }
  }
  for (int i = 0; i < kOut; ++i)
    idxOut[(size_t)m * kOut + i] = bi[i * dil];
}

// ---------------------------------------------------------------------------
// BN batch statistics: per-channel mean and rsqrt(var+eps) over M' columns.
// ---------------------------------------------------------------------------
__global__ void bn_stats(const float* __restrict__ Y, float* __restrict__ mean,
                         float* __restrict__ rvar, int Mp) {
  __shared__ float s1[256], s2[256];
  const int o = blockIdx.x;
  const float* row = Y + (size_t)o * Mp;
  float a = 0.f, b = 0.f;
  for (int i = threadIdx.x; i < Mp; i += 256) { float v = row[i]; a += v; b += v * v; }
  s1[threadIdx.x] = a; s2[threadIdx.x] = b;
  __syncthreads();
  for (int st = 128; st > 0; st >>= 1) {
    if (threadIdx.x < st) {
      s1[threadIdx.x] += s1[threadIdx.x + st];
      s2[threadIdx.x] += s2[threadIdx.x + st];
    }
    __syncthreads();
  }
  if (threadIdx.x == 0) {
    float mu = s1[0] / Mp;
    float var = s2[0] / Mp - mu * mu;
    mean[o] = mu;
    rvar[o] = rsqrtf(var + BN_EPS);
  }
}

// BN + ReLU + max over k (EdgeConv epilogue). Writes point-major f32 slab.
__global__ void bn_apply_maxk(const float* __restrict__ Y, const float* g,
                              const float* be, const float* mean, const float* rvar,
                              float* __restrict__ out, int O, int M, int Kk,
                              int outStride, int colOff) {
  const int gid = blockIdx.x * 256 + threadIdx.x;
  if (gid >= O * M) return;
  const int o = gid / M, m = gid % M;
  const float sc = g[o] * rvar[o];
  const float sh = be[o] - mean[o] * sc;
  const float* row = Y + (size_t)o * M * Kk + (size_t)m * Kk;
  float v = -3.4e38f;
  for (int kk = 0; kk < Kk; ++kk)
    v = fmaxf(v, fmaxf(row[kk] * sc + sh, 0.f));
  out[(size_t)m * outStride + colOff + o] = v;
}

// BN + ReLU in place on channel-major [O, M] (fusion conv, kept for pooling).
__global__ void bn_apply_inplace(float* __restrict__ Y, const float* g,
                                 const float* be, const float* mean,
                                 const float* rvar, int O, int M) {
  const int gid = blockIdx.x * 256 + threadIdx.x;
  if (gid >= O * M) return;
  const int o = gid / M;
  const float sc = g[o] * rvar[o];
  const float sh = be[o] - mean[o] * sc;
  Y[gid] = fmaxf(Y[gid] * sc + sh, 0.f);
}

// BN + ReLU, repack channel-major f32 -> point-major f16 (next GEMM input).
__global__ void bn_apply_to_f16(const float* __restrict__ Y, const float* g,
                                const float* be, const float* mean,
                                const float* rvar, _Float16* __restrict__ out,
                                int O, int M) {
  const int gid = blockIdx.x * 256 + threadIdx.x;
  if (gid >= O * M) return;
  const int o = gid / M, m = gid % M;
  const float sc = g[o] * rvar[o];
  const float sh = be[o] - mean[o] * sc;
  out[(size_t)m * O + o] = (_Float16)fmaxf(Y[(size_t)o * M + m] * sc + sh, 0.f);
}

// Global max-pool over N per (channel, batch).
__global__ void maxpool_n(const float* __restrict__ Yf, float* __restrict__ pooled,
                          int M, int Nn) {
  __shared__ float s[256];
  const int c = blockIdx.x, b = blockIdx.y;
  const float* row = Yf + (size_t)c * M + (size_t)b * Nn;
  float v = -3.4e38f;
  for (int i = threadIdx.x; i < Nn; i += 256) v = fmaxf(v, row[i]);
  s[threadIdx.x] = v;
  __syncthreads();
  for (int st = 128; st > 0; st >>= 1) {
    if (threadIdx.x < st) s[threadIdx.x] = fmaxf(s[threadIdx.x], s[threadIdx.x + st]);
    __syncthreads();
  }
  if (threadIdx.x == 0) pooled[c * gridDim.y + b] = s[0];
}

// ---------------------------------------------------------------------------
// Feature builders / gathers
// ---------------------------------------------------------------------------
// inputs layout [B, 3, N]; extract xy -> point-major [M, 2] for head KNN.
__global__ void extract_xy(const float* __restrict__ inp, float* __restrict__ xy, int Nn) {
  const int m = blockIdx.x * 256 + threadIdx.x;
  if (m >= 2 * Nn) return;
  const int b = m / Nn, n = m % Nn;
  xy[m * 2 + 0] = inp[b * 3 * Nn + 0 * Nn + n];
  xy[m * 2 + 1] = inp[b * 3 * Nn + 1 * Nn + n];
}

// Head EdgeConv features: [x_i(3), x_j - x_i(3), 0-pad] -> f16 [M*K, 32].
__global__ void gather_head(const float* __restrict__ inp, const int* __restrict__ idx,
                            _Float16* __restrict__ out, int Nn, int Kk, int total) {
  const int gid = blockIdx.x * 256 + threadIdx.x;
  if (gid >= total) return;
  const int c  = gid & 31;
  const int mp = gid >> 5;
  const int kk = mp % Kk;
  const int m  = mp / Kk;
  const int b  = m / Nn, n = m % Nn;
  float v = 0.f;
  if (c < 3) {
    v = inp[b * 3 * Nn + c * Nn + n];
  } else if (c < 6) {
    const int cc = c - 3;
    const int j  = idx[(size_t)m * Kk + kk];
    v = inp[b * 3 * Nn + cc * Nn + j] - inp[b * 3 * Nn + cc * Nn + n];
  }
  out[gid] = (_Float16)v;
}

// Generic EdgeConv features: [x_i(C), x_j - x_i(C)] -> f16 [M*K, 2C].
__global__ void gather_edge(const float* __restrict__ x, const int* __restrict__ idx,
                            _Float16* __restrict__ out, int C, int Nn, int Kk, int total) {
  const int gid = blockIdx.x * 256 + threadIdx.x;
  if (gid >= total) return;
  const int Cp = 2 * C;
  const int c  = gid % Cp;
  const int mp = gid / Cp;
  const int kk = mp % Kk;
  const int m  = mp / Kk;
  const int b  = m / Nn;
  float v;
  if (c < C) {
    v = x[(size_t)m * C + c];
  } else {
    const int j  = idx[(size_t)m * Kk + kk];
    const int jm = b * Nn + (j % Nn);
    const int cc = c - C;
    v = x[(size_t)jm * C + cc] - x[(size_t)m * C + cc];
  }
  out[gid] = (_Float16)v;
}

// Copy f32 [M, C] into a wider point-major slab at column offset.
__global__ void scatter_cols_f32(const float* __restrict__ src, float* __restrict__ dst,
                                 int M, int C, int dstStride, int colOff) {
  const int gid = blockIdx.x * 256 + threadIdx.x;
  if (gid >= M * C) return;
  const int m = gid / C, c = gid % C;
  dst[(size_t)m * dstStride + colOff + c] = src[gid];
}

// feats = cat(f0, f1, f2) with f1=[e1,f0], f2=[e2,e1,f0]:
// cols [0:64)=f0 [64:128)=e1 [128:192)=f0 [192:256)=e2 [256:320)=e1 [320:384)=f0
__global__ void build_feats(const float* __restrict__ f0, const float* __restrict__ f1,
                            const float* __restrict__ e2, _Float16* __restrict__ feats,
                            int M) {
  const int gid = blockIdx.x * 256 + threadIdx.x;
  if (gid >= M * 384) return;
  const int m = gid / 384, c = gid % 384;
  float v;
  if      (c <  64) v = f0[(size_t)m *  64 + c];
  else if (c < 128) v = f1[(size_t)m * 128 + (c -  64)];   // e1 = f1[:, :64]
  else if (c < 192) v = f0[(size_t)m *  64 + (c - 128)];
  else if (c < 256) v = e2[(size_t)m *  64 + (c - 192)];
  else if (c < 320) v = f1[(size_t)m * 128 + (c - 256)];
  else              v = f0[(size_t)m *  64 + (c - 320)];
  feats[gid] = (_Float16)v;
}

// xcat = cat(broadcast(pooled)[1024], feats[384]) -> f16 [M, 1408]
__global__ void build_xcat(const float* __restrict__ pooled,
                           const _Float16* __restrict__ feats,
                           _Float16* __restrict__ xcat, int M, int Nn) {
  const int gid = blockIdx.x * 256 + threadIdx.x;
  if (gid >= M * 1408) return;
  const int m = gid / 1408, c = gid % 1408;
  const int b = m / Nn;
  xcat[gid] = (c < 1024) ? (_Float16)pooled[c * 2 + b]
                         : feats[(size_t)m * 384 + (c - 1024)];
}

// Pack f32 weights [O, Cin] -> f16 [O, Cpad] (zero pad).
__global__ void pack_w_f16(const float* __restrict__ w, _Float16* __restrict__ out,
                           int O, int Cin, int Cpad) {
  const int gid = blockIdx.x * 256 + threadIdx.x;
  if (gid >= O * Cpad) return;
  const int o = gid / Cpad, c = gid % Cpad;
  out[gid] = (c < Cin) ? (_Float16)w[(size_t)o * Cin + c] : (_Float16)0.f;
}

// Final 1x1 conv (no norm/act): out[m] = wc . x[m,:] + bc
__global__ void final_conv(const _Float16* __restrict__ xb, const float* __restrict__ wc,
                           const float* __restrict__ bc, float* __restrict__ out, int M) {
  const int m = blockIdx.x * 256 + threadIdx.x;
  if (m >= M) return;
  float a = bc[0];
  const _Float16* r = xb + (size_t)m * 256;
  for (int c = 0; c < 256; ++c) a += wc[c] * (float)r[c];
  out[m] = a;
}

// ===========================================================================
extern "C" void kernel_launch(void* const* d_in, const int* in_sizes, int n_in,
                              void* d_out, int out_size, void* d_ws, size_t ws_size,
                              hipStream_t stream) {
  (void)in_sizes; (void)n_in; (void)out_size; (void)ws_size;

  const float* inp = (const float*)d_in[0];
  const float* wh = (const float*)d_in[1];
  const float* gh = (const float*)d_in[3];  const float* eh = (const float*)d_in[4];
  const float* w1 = (const float*)d_in[5];
  const float* g1 = (const float*)d_in[7];  const float* e1 = (const float*)d_in[8];
  const float* w2 = (const float*)d_in[9];
  const float* g2 = (const float*)d_in[11]; const float* e2g = (const float*)d_in[12];
  const float* wf = (const float*)d_in[13];
  const float* gf = (const float*)d_in[15]; const float* ef = (const float*)d_in[16];
  const float* wa = (const float*)d_in[17];
  const float* ga = (const float*)d_in[19]; const float* ea = (const float*)d_in[20];
  const float* wb = (const float*)d_in[21];
  const float* gb = (const float*)d_in[23]; const float* eb = (const float*)d_in[24];
  const float* wc = (const float*)d_in[25]; const float* bc = (const float*)d_in[26];
  float* outp = (float*)d_out;

  // ---- workspace arena (deterministic offsets) ----
  char* base = (char*)d_ws;
  size_t off = 0;
  auto arena = [&](size_t bytes) -> char* {
    char* p = base + off;
    off += (bytes + 255) & ~(size_t)255;
    return p;
  };
  const int MK = MT * KNN;                       // 81920
  int*       idx0   = (int*)      arena((size_t)MK * 4);
  int*       idx1   = (int*)      arena((size_t)MK * 4);
  int*       idx2   = (int*)      arena((size_t)MK * 4);
  float*     xy     = (float*)    arena((size_t)MT * 2 * 4);
  _Float16*  wh16   = (_Float16*) arena((size_t)64 * 32 * 2);
  _Float16*  w116   = (_Float16*) arena((size_t)64 * 128 * 2);
  _Float16*  w216   = (_Float16*) arena((size_t)64 * 256 * 2);
  _Float16*  wf16   = (_Float16*) arena((size_t)1024 * 384 * 2);
  _Float16*  wa16   = (_Float16*) arena((size_t)512 * 1408 * 2);
  _Float16*  wb16   = (_Float16*) arena((size_t)256 * 512 * 2);
  float*     mean   = (float*)    arena(1024 * 4);
  float*     rvar   = (float*)    arena(1024 * 4);
  float*     pooled = (float*)    arena(1024 * 2 * 4);
  float*     Y      = (float*)    arena((size_t)1024 * MT * 4);      // shared GEMM out
  _Float16*  feat   = (_Float16*) arena((size_t)MT * 1408 * 2);      // featH/1/2 + xcat
  _Float16*  feats16= (_Float16*) arena((size_t)MT * 384 * 2);
  float*     f0     = (float*)    arena((size_t)MT * 64 * 4);
  float*     f1     = (float*)    arena((size_t)MT * 128 * 4);
  float*     e2b    = (float*)    arena((size_t)MT * 64 * 4);
  _Float16*  xa16   = (_Float16*) arena((size_t)MT * 512 * 2);
  _Float16*  xb16   = (_Float16*) arena((size_t)MT * 256 * 2);

  auto blks = [](long n) { return (unsigned)((n + 255) / 256); };
  auto wtile = [](int C) { return (size_t)16 * C * sizeof(_Float16); };  // LDS bytes

  // ---- pack weights to f16 (head padded 6 -> 32) ----
  pack_w_f16<<<blks(64 * 32),    256, 0, stream>>>(wh, wh16, 64, 6, 32);
  pack_w_f16<<<blks(64 * 128),   256, 0, stream>>>(w1, w116, 64, 128, 128);
  pack_w_f16<<<blks(64 * 256),   256, 0, stream>>>(w2, w216, 64, 256, 256);
  pack_w_f16<<<blks(1024 * 384), 256, 0, stream>>>(wf, wf16, 1024, 384, 384);
  pack_w_f16<<<blks(512 * 1408), 256, 0, stream>>>(wa, wa16, 512, 1408, 1408);
  pack_w_f16<<<blks(256 * 512),  256, 0, stream>>>(wb, wb16, 256, 512, 512);

  dim3 knnGrid(NP / 64, BB);

  // ---- head: KNN on xy, EdgeConv(3 -> 64) ----
  extract_xy<<<blks(MT), 256, 0, stream>>>(inp, xy, NP);
  knn_kernel<5, 2><<<knnGrid, 64, 0, stream>>>(xy, idx0, 2, NP, KNN, 1);
  gather_head<<<blks((long)MK * 32), 256, 0, stream>>>(inp, idx0, feat, NP, KNN, MK * 32);
  wmma_gemm_f16<<<dim3(MK / 512, 64 / 16), 256, wtile(32), stream>>>(wh16, feat, Y, 64, 32, MK);
  bn_stats<<<64, 256, 0, stream>>>(Y, mean, rvar, MK);
  bn_apply_maxk<<<blks((long)64 * MT), 256, 0, stream>>>(Y, gh, eh, mean, rvar,
                                                         f0, 64, MT, KNN, 64, 0);

  // ---- dense block 1: KNN on f0, EdgeConv(64 -> 64), f1 = [e1, f0] ----
  knn_kernel<10, 64><<<knnGrid, 64, 0, stream>>>(f0, idx1, 64, NP, KNN, 2);
  gather_edge<<<blks((long)MK * 128), 256, 0, stream>>>(f0, idx1, feat, 64, NP, KNN, MK * 128);
  wmma_gemm_f16<<<dim3(MK / 512, 64 / 16), 256, wtile(128), stream>>>(w116, feat, Y, 64, 128, MK);
  bn_stats<<<64, 256, 0, stream>>>(Y, mean, rvar, MK);
  bn_apply_maxk<<<blks((long)64 * MT), 256, 0, stream>>>(Y, g1, e1, mean, rvar,
                                                         f1, 64, MT, KNN, 128, 0);
  scatter_cols_f32<<<blks((long)MT * 64), 256, 0, stream>>>(f0, f1, MT, 64, 128, 64);

  // ---- dense block 2: KNN on f1, EdgeConv(128 -> 64) ----
  knn_kernel<15, 128><<<knnGrid, 64, 0, stream>>>(f1, idx2, 128, NP, KNN, 3);
  gather_edge<<<blks((long)MK * 256), 256, 0, stream>>>(f1, idx2, feat, 128, NP, KNN, MK * 256);
  wmma_gemm_f16<<<dim3(MK / 512, 64 / 16), 256, wtile(256), stream>>>(w216, feat, Y, 64, 256, MK);
  bn_stats<<<64, 256, 0, stream>>>(Y, mean, rvar, MK);
  bn_apply_maxk<<<blks((long)64 * MT), 256, 0, stream>>>(Y, g2, e2g, mean, rvar,
                                                         e2b, 64, MT, KNN, 64, 0);

  // ---- fusion conv 384 -> 1024, BN+ReLU, global max pool ----
  build_feats<<<blks((long)MT * 384), 256, 0, stream>>>(f0, f1, e2b, feats16, MT);
  wmma_gemm_f16<<<dim3(MT / 512, 1024 / 16), 256, wtile(384), stream>>>(wf16, feats16, Y, 1024, 384, MT);
  bn_stats<<<1024, 256, 0, stream>>>(Y, mean, rvar, MT);
  bn_apply_inplace<<<blks((long)1024 * MT), 256, 0, stream>>>(Y, gf, ef, mean, rvar, 1024, MT);
  maxpool_n<<<dim3(1024, BB), 256, 0, stream>>>(Y, pooled, MT, NP);

  // ---- prediction head: 1408 -> 512 -> 256 -> 1 ----
  build_xcat<<<blks((long)MT * 1408), 256, 0, stream>>>(pooled, feats16, feat, MT, NP);
  wmma_gemm_f16<<<dim3(MT / 512, 512 / 16), 256, wtile(1408), stream>>>(wa16, feat, Y, 512, 1408, MT);
  bn_stats<<<512, 256, 0, stream>>>(Y, mean, rvar, MT);
  bn_apply_to_f16<<<blks((long)512 * MT), 256, 0, stream>>>(Y, ga, ea, mean, rvar, xa16, 512, MT);

  wmma_gemm_f16<<<dim3(MT / 512, 256 / 16), 256, wtile(512), stream>>>(wb16, xa16, Y, 256, 512, MT);
  bn_stats<<<256, 256, 0, stream>>>(Y, mean, rvar, MT);
  bn_apply_to_f16<<<blks((long)256 * MT), 256, 0, stream>>>(Y, gb, eb, mean, rvar, xb16, 256, MT);

  final_conv<<<blks(MT), 256, 0, stream>>>(xb16, wc, bc, outp, MT);
}
